// Encoder_meta_1151051235834
// MI455X (gfx1250) — matrix operand order repaired and checked by reading, not compiled, not run on hardware
//
#include <hip/hip_runtime.h>
#include <math.h>

// ---------------- problem constants ----------------
#define BATCH 256
#define TSEQ  512
#define HID   512
#define XPDIM 256
#define NWG   32
#define NTHR  256   // 8 wave32 waves per workgroup

typedef __attribute__((ext_vector_type(16))) __bf16 v16bf;
typedef __attribute__((ext_vector_type(8)))  float  v8f;
typedef __attribute__((ext_vector_type(4)))  unsigned int v4u;
typedef __attribute__((ext_vector_type(8)))  int v8i;
typedef __attribute__((ext_vector_type(4)))  int v4i;

// float -> bf16 (round to nearest even)
__device__ __forceinline__ __bf16 f2bf(float f) {
    union { float f; unsigned u; } c; c.f = f;
    unsigned r = (c.u + 0x7FFFu + ((c.u >> 16) & 1u)) >> 16;
    union { unsigned short s; __bf16 b; } o; o.s = (unsigned short)r;
    return o.b;
}

__device__ __forceinline__ v8f wmma_bf16(v16bf a, v16bf b, v8f c) {
    return __builtin_amdgcn_wmma_f32_16x16x32_bf16(
        false, a, false, b, (short)0, c, false, false);
}

// ---- A fragment (16x32 bf16, MxK) from row-major bf16 [M][ldK] ----
__device__ __forceinline__ v16bf loadA_bf16(const __bf16* src, int ldK,
                                            int m0, int k0, int lane) {
    v16bf a;
    const __bf16* row = src + (size_t)(m0 + (lane & 15)) * ldK
                            + k0 + ((lane & 16) ? 8 : 0);
#pragma unroll
    for (int v = 0; v < 8; ++v) {
        int k = ((v & 4) ? 16 : 0) + 2 * (v & 3);
        union { unsigned u; __bf16 h[2]; } p;
        p.u = *(const unsigned*)(row + k);
        a[2 * v]     = p.h[0];
        a[2 * v + 1] = p.h[1];
    }
    return a;
}

// ---- A fragment from row-major fp32 source (convert on the fly) ----
__device__ __forceinline__ v16bf loadA_f32(const float* base, size_t rowStride,
                                           int m0, int lane) {
    v16bf a;
    const float* p = base + (size_t)(m0 + (lane & 15)) * rowStride
                          + ((lane & 16) ? 8 : 0);
#pragma unroll
    for (int v = 0; v < 8; ++v) {
        int k = ((v & 4) ? 16 : 0) + 2 * (v & 3);
        a[2 * v]     = f2bf(p[k]);
        a[2 * v + 1] = f2bf(p[k + 1]);
    }
    return a;
}

// ---- B fragment: 512 contiguous bf16, lane-contiguous 32B ----
__device__ __forceinline__ v16bf loadB(const __bf16* fragBase, int lane) {
    return *(const v16bf*)(fragBase + lane * 16);
}

// ---- TDM: DMA 3 gate fragments (3 rows x 1024B, stride 32768B) into LDS ----
// D# built per CDNA5 ISA ch.8: group0 = {count/type/addr}, group1 = dims/strides.
__device__ __forceinline__ void tdm_load_3frag(const __bf16* gsrc, unsigned lds_off) {
    unsigned long long ga = (unsigned long long)(uintptr_t)gsrc;
    v4u g0 = { 1u,                                   // count=1, user descriptor
               lds_off,                              // lds_addr (bytes)
               (unsigned)ga,                         // global_addr[31:0]
               (unsigned)((ga >> 32) & 0x1FFFFFFu) | 0x80000000u }; // [56:32] | type=2
    v8i g1 = { 0,                                    // wg_mask=0, data_size=1B
               (int)(1024u << 16),                   // tensor_dim0 = 1024 (lo16)
               (int)(3u << 16),                      // tensor_dim1 = 3
               (int)(1024u << 16),                   // tile_dim0   = 1024
               3,                                    // tile_dim1=3, tile_dim2=0
               32768,                                // tensor_dim0_stride = 32KB
               0, 0 };
    v4i z4 = { 0, 0, 0, 0 };
#if defined(__clang_major__) && (__clang_major__ >= 23)
    v8i z8 = { 0, 0, 0, 0, 0, 0, 0, 0 };
    __builtin_amdgcn_tensor_load_to_lds(g0, g1, z4, z4, z8, 0);
#else
    __builtin_amdgcn_tensor_load_to_lds(g0, g1, z4, z4, 0);
#endif
}

// ---- grid-wide barrier (sense-reversal, replay-safe: only cnt must start 0) ----
__device__ __forceinline__ void gbar(unsigned* cnt, unsigned* gen) {
    __syncthreads();
    __threadfence();
    if (threadIdx.x == 0) {
        unsigned g = *(volatile unsigned*)gen;
        if (atomicAdd(cnt, 1u) == NWG - 1u) {
            *(volatile unsigned*)cnt = 0u;
            __threadfence();
            atomicAdd(gen, 1u);
        } else {
            while (*(volatile unsigned*)gen == g) __builtin_amdgcn_s_sleep(1);
        }
    }
    __threadfence();
    __syncthreads();
}

// ---- pack W (N x K fp32, row-major) into WMMA B fragments of W^T (K x N) ----
__global__ void pack_wT(const float* __restrict__ W, __bf16* __restrict__ dst,
                        int K, int N) {
    int gid = blockIdx.x * blockDim.x + threadIdx.x;
    if (gid >= K * N) return;
    int frag = gid >> 9, r = gid & 511, lane = r >> 4, e = r & 15;
    int NT = N >> 4;
    int kt = frag / NT, nt = frag - kt * NT;
    int k = kt * 32 + ((lane & 16) ? 8 : 0) + (e & 7) + ((e & 8) ? 16 : 0);
    int n = nt * 16 + (lane & 15);
    dst[gid] = f2bf(W[(size_t)n * K + k]);
}

// ---------------- persistent cooperative GRU kernel ----------------
__global__ __launch_bounds__(NTHR) void gru_persistent(
    const float* __restrict__ xl, const float* __restrict__ xti,
    const float* __restrict__ xw, const float* __restrict__ xs,
    const float* __restrict__ b_ih, const float* __restrict__ b_hh,
    const float* __restrict__ bt,
    float* h32_0, float* h32_1, __bf16* h16_0, __bf16* h16_1, __bf16* xp,
    const __bf16* __restrict__ WpTh, const __bf16* __restrict__ WpTx,
    const __bf16* __restrict__ WpIh, const __bf16* __restrict__ WpHh,
    unsigned* cnt, unsigned* gen) {

    // double-buffered staging for 3 gate B-fragments (2 x 3KB); sole LDS
    // object in this kernel -> LDS offset 0 (TDM descriptor assumption).
    __shared__ __align__(128) __bf16 stage[2 * 3 * 512];

    const int lane = threadIdx.x & 31;
    const int w8   = threadIdx.x >> 5;                       // wave in WG: 0..7
    const int wid  = (blockIdx.x << 3) + w8;                 // 0..255
    const bool isW0 = (threadIdx.x < 32);

    for (int t = 0; t < TSEQ; ++t) {
        const float*  h32c = (t & 1) ? h32_1 : h32_0;
        float*        h32n = (t & 1) ? h32_0 : h32_1;
        const __bf16* h16c = (t & 1) ? h16_1 : h16_0;
        __bf16*       h16n = (t & 1) ? h16_0 : h16_1;

        // ======== Phase A: x' = tanh(h @ Wt_h^T + x_t @ Wt_x^T + bt) (256x256)
        {
            int mt = wid >> 4, nt = wid & 15;
            v8f acc = {};
#pragma unroll 4
            for (int kt = 0; kt < 16; ++kt) {
                __builtin_prefetch(WpTh + ((((kt + 1) & 15) * 16 + nt) << 9), 0, 1);
                v16bf a = loadA_bf16(h16c, HID, mt * 16, kt * 32, lane);
                v16bf b = loadB(WpTh + ((kt * 16 + nt) << 9), lane);
                acc = wmma_bf16(a, b, acc);
            }
#pragma unroll
            for (int kt = 0; kt < 5; ++kt) {
                const float* xbase; size_t stride;
                if      (kt == 0) { xbase = xl  + (size_t)t * 64;      stride = (size_t)TSEQ * 64; }
                else if (kt == 1) { xbase = xl  + (size_t)t * 64 + 32; stride = (size_t)TSEQ * 64; }
                else if (kt == 2) { xbase = xti + (size_t)t * 32;      stride = (size_t)TSEQ * 32; }
                else if (kt == 3) { xbase = xw  + (size_t)t * 32;      stride = (size_t)TSEQ * 32; }
                else              { xbase = xs  + (size_t)t * 32;      stride = (size_t)TSEQ * 32; }
                v16bf a = loadA_f32(xbase, stride, mt * 16, lane);
                v16bf b = loadB(WpTx + ((kt * 16 + nt) << 9), lane);
                acc = wmma_bf16(a, b, acc);
            }
            int n  = nt * 16 + (lane & 15);
            int mb = mt * 16 + ((lane & 16) ? 8 : 0);
            float bias = bt[n];
#pragma unroll
            for (int r = 0; r < 8; ++r)
                xp[(size_t)(mb + r) * XPDIM + n] = f2bf(tanhf(acc[r] + bias));
        }
        gbar(cnt, gen);

        // ======== Phase B: gates + h_new (256x512)
        // WG owns column tile nt2 = blockIdx.x; wave handles mt0 = w8, mt1 = w8+8,
        // sharing TDM-staged B fragments across both M-subtiles.
        {
            const int nt2 = blockIdx.x;
            const int mt0 = w8, mt1 = w8 + 8;
            v8f aR0 = {}, aZ0 = {}, aN0 = {}, hR0 = {}, hZ0 = {}, hN0 = {};
            v8f aR1 = {}, aZ1 = {}, aN1 = {}, hR1 = {}, hZ1 = {}, hN1 = {};

            // kt 0..7  : x' @ W_ih^T (K=256);  kt 8..23 : h @ W_hh^T (K=512)
            if (isW0)
                tdm_load_3frag(WpIh + ((0 * 96 + nt2) << 9), 0u);

            for (int kt = 0; kt < 24; ++kt) {
                if (isW0) {
                    if (kt + 1 < 24) {
                        const __bf16* nsrc = (kt + 1 < 8)
                            ? WpIh + (((kt + 1) * 96 + nt2) << 9)
                            : WpHh + (((kt + 1 - 8) * 96 + nt2) << 9);
                        tdm_load_3frag(nsrc, ((kt + 1) & 1) ? 3072u : 0u);
                        __builtin_amdgcn_s_wait_tensorcnt(1);   // oldest done
                    } else {
                        __builtin_amdgcn_s_wait_tensorcnt(0);
                    }
                }
                __syncthreads();                 // staged buf[kt&1] visible

                // The TDM engine wrote `stage` behind the compiler's back;
                // force it to assume LDS was modified so the ds_load_b128
                // reads below are not folded away as loads of undef.
                const __bf16* bb = stage + ((kt & 1) ? 3 * 512 : 0);
                asm volatile("" : "+v"(bb) : : "memory");

                v16bf bR = loadB(bb,        lane);    // ds_load_b128 x2
                v16bf bZ = loadB(bb + 512,  lane);
                v16bf bN = loadB(bb + 1024, lane);
                v16bf a0, a1;
                if (kt < 8) {
                    a0 = loadA_bf16(xp, XPDIM, mt0 * 16, kt * 32, lane);
                    a1 = loadA_bf16(xp, XPDIM, mt1 * 16, kt * 32, lane);
                    aR0 = wmma_bf16(a0, bR, aR0);  aR1 = wmma_bf16(a1, bR, aR1);
                    aZ0 = wmma_bf16(a0, bZ, aZ0);  aZ1 = wmma_bf16(a1, bZ, aZ1);
                    aN0 = wmma_bf16(a0, bN, aN0);  aN1 = wmma_bf16(a1, bN, aN1);
                } else {
                    a0 = loadA_bf16(h16c, HID, mt0 * 16, (kt - 8) * 32, lane);
                    a1 = loadA_bf16(h16c, HID, mt1 * 16, (kt - 8) * 32, lane);
                    hR0 = wmma_bf16(a0, bR, hR0);  hR1 = wmma_bf16(a1, bR, hR1);
                    hZ0 = wmma_bf16(a0, bZ, hZ0);  hZ1 = wmma_bf16(a1, bZ, hZ1);
                    hN0 = wmma_bf16(a0, bN, hN0);  hN1 = wmma_bf16(a1, bN, hN1);
                }
                __syncthreads();                 // all consumed before overwrite
            }

            int n  = nt2 * 16 + (lane & 15);
            float biR = b_ih[n], biZ = b_ih[n + HID], biN = b_ih[n + 2 * HID];
            float bhR = b_hh[n], bhZ = b_hh[n + HID], bhN = b_hh[n + 2 * HID];
#pragma unroll
            for (int half = 0; half < 2; ++half) {
                int mb = (half ? mt1 : mt0) * 16 + ((lane & 16) ? 8 : 0);
                const v8f& AR = half ? aR1 : aR0; const v8f& AZ = half ? aZ1 : aZ0;
                const v8f& AN = half ? aN1 : aN0; const v8f& HR = half ? hR1 : hR0;
                const v8f& HZ = half ? hZ1 : hZ0; const v8f& HN = half ? hN1 : hN0;
#pragma unroll
                for (int r = 0; r < 8; ++r) {
                    float hold = h32c[(size_t)(mb + r) * HID + n];
                    float rg = 1.f / (1.f + __expf(-(AR[r] + biR + HR[r] + bhR)));
                    float zg = 1.f / (1.f + __expf(-(AZ[r] + biZ + HZ[r] + bhZ)));
                    float ng = tanhf(AN[r] + biN + rg * (HN[r] + bhN));
                    float hv = (1.f - zg) * ng + zg * hold;
                    h32n[(size_t)(mb + r) * HID + n] = hv;
                    h16n[(size_t)(mb + r) * HID + n] = f2bf(hv);
                }
            }
        }
        gbar(cnt, gen);
    }
}

__global__ void copy_out(const float* __restrict__ src, float* __restrict__ dst, int n) {
    int i = blockIdx.x * blockDim.x + threadIdx.x;
    if (i < n) dst[i] = src[i];
}

extern "C" void kernel_launch(void* const* d_in, const int* in_sizes, int n_in,
                              void* d_out, int out_size, void* d_ws, size_t ws_size,
                              hipStream_t stream) {
    (void)in_sizes; (void)n_in; (void)out_size; (void)ws_size;

    const float* xl   = (const float*)d_in[0];
    const float* xti  = (const float*)d_in[1];
    const float* xw   = (const float*)d_in[2];
    const float* xs   = (const float*)d_in[3];
    const float* W_ih = (const float*)d_in[4];
    const float* W_hh = (const float*)d_in[5];
    const float* b_ih = (const float*)d_in[6];
    const float* b_hh = (const float*)d_in[7];
    const float* Wt_h = (const float*)d_in[8];
    const float* Wt_x = (const float*)d_in[9];
    const float* bt   = (const float*)d_in[10];

    char* ws = (char*)d_ws;
    size_t off = 0;
    auto carve = [&](size_t bytes) {
        char* p = ws + off;
        off = (off + bytes + 255) & ~(size_t)255;
        return p;
    };
    unsigned* ctl   = (unsigned*)carve(256);                 // [0]=cnt [1]=gen
    float*    h32_0 = (float*)   carve((size_t)BATCH * HID * 4);
    float*    h32_1 = (float*)   carve((size_t)BATCH * HID * 4);
    __bf16*   h16_0 = (__bf16*)  carve((size_t)BATCH * HID * 2);
    __bf16*   h16_1 = (__bf16*)  carve((size_t)BATCH * HID * 2);
    size_t zeroBytes = off;                                  // ctl + h buffers
    __bf16*   xp    = (__bf16*)  carve((size_t)BATCH * XPDIM * 2);
    __bf16*   WpTh  = (__bf16*)  carve((size_t)HID * XPDIM * 2);
    __bf16*   WpTx  = (__bf16*)  carve((size_t)160 * XPDIM * 2);
    __bf16*   WpIh  = (__bf16*)  carve((size_t)XPDIM * 3 * HID * 2);
    __bf16*   WpHh  = (__bf16*)  carve((size_t)HID * 3 * HID * 2);

    hipMemsetAsync(d_ws, 0, zeroBytes, stream);

    pack_wT<<<(512 * 256 + 255) / 256, 256, 0, stream>>>(Wt_h, WpTh, 512, 256);
    pack_wT<<<(160 * 256 + 255) / 256, 256, 0, stream>>>(Wt_x, WpTx, 160, 256);
    pack_wT<<<(256 * 1536 + 255) / 256, 256, 0, stream>>>(W_ih, WpIh, 256, 1536);
    pack_wT<<<(512 * 1536 + 255) / 256, 256, 0, stream>>>(W_hh, WpHh, 512, 1536);

    gru_persistent<<<NWG, NTHR, 0, stream>>>(
        xl, xti, xw, xs, b_ih, b_hh, bt,
        h32_0, h32_1, h16_0, h16_1, xp,
        WpTh, WpTx, WpIh, WpHh,
        &ctl[0], &ctl[1]);

    copy_out<<<(BATCH * HID + 255) / 256, 256, 0, stream>>>(
        h32_0, (float*)d_out, BATCH * HID);
}
